// QuantizedLinearLayer_35768487641515
// MI455X (gfx1250) — 8-bit matrix operand order repaired and checked by reading, compiled, not run on hardware
//
#include <hip/hip_runtime.h>
#include <stdint.h>

typedef int v8i __attribute__((ext_vector_type(8)));
typedef int vec4i __attribute__((vector_size(16)));

static constexpr int Mdim = 8192;
static constexpr int Ndim = 4096;
static constexpr int Kdim = 4096;

static constexpr int BM = 64;
static constexpr int BN = 256;
static constexpr int BK = 128;
static constexpr int A_ST = BK + 16;
static constexpr int B_ST = BK + 16;

#if defined(__has_builtin)
#if __has_builtin(__builtin_amdgcn_global_load_async_to_lds_b128)
#define ASYNC_LDS_BUILTIN 1
#endif
#if __has_builtin(__builtin_amdgcn_s_wait_asynccnt)
#define WAIT_ASYNC_BUILTIN 1
#endif
#endif

__device__ __forceinline__ void async_copy_b128(const void* g, void* l) {
#if defined(ASYNC_LDS_BUILTIN)
  __builtin_amdgcn_global_load_async_to_lds_b128(
      (__attribute__((address_space(1))) vec4i*)g,
      (__attribute__((address_space(3))) vec4i*)l, 0, 0);
#else
  auto l3 = (__attribute__((address_space(3))) vec4i*)l;
  asm volatile("global_load_async_to_lds_b128 %0, %1, off"
               :: "v"(l3), "v"(g) : "memory");
#endif
}

template <int N>
__device__ __forceinline__ void wait_asynccnt() {
#if defined(WAIT_ASYNC_BUILTIN)
  __builtin_amdgcn_s_wait_asynccnt(N);
#else
  asm volatile("s_wait_asynccnt %0" :: "n"(N) : "memory");
#endif
}

__device__ __forceinline__ int quant1(float v, float inv) {
  float t = fminf(fmaxf(v * inv, -128.0f), 127.0f);
  return __float2int_rn(t);
}

__global__ void __launch_bounds__(256)
quantize_kernel(const float* __restrict__ x, uint32_t* __restrict__ xq,
                const float* __restrict__ act) {
  const float inv = 1.0f / act[0];
  const size_t i = (size_t)blockIdx.x * 256 + threadIdx.x;
  const float4 v = reinterpret_cast<const float4*>(x)[i];
  uint32_t p = (uint32_t)(quant1(v.x, inv) & 0xff)
             | ((uint32_t)(quant1(v.y, inv) & 0xff) << 8)
             | ((uint32_t)(quant1(v.z, inv) & 0xff) << 16)
             | ((uint32_t)(quant1(v.w, inv) & 0xff) << 24);
  xq[i] = p;
}

struct WaveCtx {
  int lane, waveM, waveN, mrow, khalf;
};

__device__ __forceinline__ void compute_tile(const int8_t* lA, const int8_t* lB,
                                             const WaveCtx& wc, v8i acc[2][4]) {
#pragma unroll
  for (int ks = 0; ks < 2; ++ks) {
    const int kk = ks * 64;

    v8i afrag[2];
#pragma unroll
    for (int mt = 0; mt < 2; ++mt) {
      const int8_t* ap =
          &lA[(wc.waveM * 32 + mt * 16 + wc.mrow) * A_ST + kk + wc.khalf * 8];
      const int2 p0 = *reinterpret_cast<const int2*>(ap + 0);
      const int2 p1 = *reinterpret_cast<const int2*>(ap + 16);
      const int2 p2 = *reinterpret_cast<const int2*>(ap + 32);
      const int2 p3 = *reinterpret_cast<const int2*>(ap + 48);
      v8i a;
      a[0] = p0.x; a[1] = p0.y; a[2] = p1.x; a[3] = p1.y;
      a[4] = p2.x; a[5] = p2.y; a[6] = p3.x; a[7] = p3.y;
      afrag[mt] = a;
    }

    v8i bfrag[4];
#pragma unroll
    for (int nt = 0; nt < 4; ++nt) {
      const int8_t* bp =
          &lB[(wc.waveN * 64 + nt * 16 + wc.mrow) * B_ST + kk + wc.khalf * 8];
      const int2 q0 = *reinterpret_cast<const int2*>(bp + 0);
      const int2 q1 = *reinterpret_cast<const int2*>(bp + 16);
      const int2 q2 = *reinterpret_cast<const int2*>(bp + 32);
      const int2 q3 = *reinterpret_cast<const int2*>(bp + 48);
      v8i b;
      b[0] = q0.x; b[1] = q0.y; b[2] = q1.x; b[3] = q1.y;
      b[4] = q2.x; b[5] = q2.y; b[6] = q3.x; b[7] = q3.y;
      bfrag[nt] = b;
    }

#pragma unroll
    for (int mt = 0; mt < 2; ++mt)
#pragma unroll
      for (int nt = 0; nt < 4; ++nt)
        acc[mt][nt] = __builtin_amdgcn_wmma_i32_16x16x64_iu8(
             true, afrag[mt],  true, bfrag[nt],
            acc[mt][nt],  false,  false);
  }
}

__device__ __forceinline__ void epilogue(const WaveCtx& wc, v8i acc[2][4],
                                         int mBlock, int nBlock,
                                         const float* scale, const float* bias,
                                         float as, float* out) {
#pragma unroll
  for (int nt = 0; nt < 4; ++nt) {
    const int n = nBlock + wc.waveN * 64 + nt * 16 + (wc.lane & 15);
    const float sc = scale[n] * as;
    const float bi = bias[n];
#pragma unroll
    for (int mt = 0; mt < 2; ++mt) {
      const int mbase = mBlock + wc.waveM * 32 + mt * 16 + wc.khalf * 8;
#pragma unroll
      for (int r = 0; r < 8; ++r) {
        out[(size_t)(mbase + r) * Ndim + n] = (float)acc[mt][nt][r] * sc + bi;
      }
    }
  }
}

__global__ void __launch_bounds__(256)
qgemm_wmma_async(const int8_t* __restrict__ xq,
                 const int8_t* __restrict__ w,
                 const float* __restrict__ scale,
                 const float* __restrict__ bias,
                 const float* __restrict__ act,
                 float* __restrict__ out) {
  __shared__ __align__(16) int8_t lA[2][BM * A_ST];
  __shared__ __align__(16) int8_t lB[2][BN * B_ST];

  const int tid = threadIdx.x;
  WaveCtx wc;
  wc.lane  = tid & 31;
  const int wave = tid >> 5;
  wc.waveM = wave & 1;
  wc.waveN = wave >> 1;
  wc.mrow  = wc.lane & 15;
  wc.khalf = wc.lane >> 4;

  const int mBlock = blockIdx.y * BM;
  const int nBlock = blockIdx.x * BN;

  auto issue_tile = [&](int buf, int k0) {
#pragma unroll
    for (int it = 0; it < 2; ++it) {
      const int idx = tid + it * 256;
      const int r = idx >> 3, c = (idx & 7) * 16;
      async_copy_b128(xq + (size_t)(mBlock + r) * Kdim + k0 + c,
                      &lA[buf][r * A_ST + c]);
    }
#pragma unroll
    for (int it = 0; it < 8; ++it) {
      const int idx = tid + it * 256;
      const int r = idx >> 3, c = (idx & 7) * 16;
      async_copy_b128(w + (size_t)(nBlock + r) * Kdim + k0 + c,
                      &lB[buf][r * B_ST + c]);
    }
  };

  v8i acc[2][4] = {};

  constexpr int NIT = Kdim / BK;
  issue_tile(0, 0);
  for (int it = 0; it < NIT; ++it) {
    const int cur = it & 1;
    if (it + 1 < NIT) {
      issue_tile(cur ^ 1, (it + 1) * BK);
      wait_asynccnt<10>();
    } else {
      wait_asynccnt<0>();
    }
    __syncthreads();
    compute_tile(lA[cur], lB[cur], wc, acc);
    __syncthreads();
  }

  epilogue(wc, acc, mBlock, nBlock, scale, bias, act[0], out);
}

__global__ void __launch_bounds__(256)
qgemm_wmma_fused(const float* __restrict__ xf, const int8_t* __restrict__ w,
                 const float* __restrict__ scale, const float* __restrict__ bias,
                 const float* __restrict__ act, float* __restrict__ out) {
  __shared__ __align__(16) int8_t lA[BM * A_ST];
  __shared__ __align__(16) int8_t lB[BN * B_ST];

  const int tid = threadIdx.x;
  WaveCtx wc;
  wc.lane  = tid & 31;
  const int wave = tid >> 5;
  wc.waveM = wave & 1;
  wc.waveN = wave >> 1;
  wc.mrow  = wc.lane & 15;
  wc.khalf = wc.lane >> 4;

  const int mBlock = blockIdx.y * BM;
  const int nBlock = blockIdx.x * BN;
  const float invAs = 1.0f / act[0];

  v8i acc[2][4] = {};

  for (int k0 = 0; k0 < Kdim; k0 += BK) {
    __syncthreads();
    {
      const int r = tid >> 2;
      const int cseg = (tid & 3) * 32;
      const float* src = xf + (size_t)(mBlock + r) * Kdim + k0 + cseg;
      uint32_t* dst = reinterpret_cast<uint32_t*>(&lA[r * A_ST + cseg]);
#pragma unroll
      for (int j = 0; j < 8; ++j) {
        const float4 v = reinterpret_cast<const float4*>(src)[j];
        dst[j] = (uint32_t)(quant1(v.x, invAs) & 0xff)
               | ((uint32_t)(quant1(v.y, invAs) & 0xff) << 8)
               | ((uint32_t)(quant1(v.z, invAs) & 0xff) << 16)
               | ((uint32_t)(quant1(v.w, invAs) & 0xff) << 24);
      }
    }
#pragma unroll
    for (int it = 0; it < 8; ++it) {
      const int idx = tid + it * 256;
      const int r = idx >> 3, c = (idx & 7) * 16;
      const int4 v = *reinterpret_cast<const int4*>(
          w + (size_t)(nBlock + r) * Kdim + k0 + c);
      *reinterpret_cast<int4*>(&lB[r * B_ST + c]) = v;
    }
    __syncthreads();
    compute_tile(lA, lB, wc, acc);
  }

  epilogue(wc, acc, mBlock, nBlock, scale, bias, act[0], out);
}

extern "C" void kernel_launch(void* const* d_in, const int* in_sizes, int n_in,
                              void* d_out, int out_size, void* d_ws, size_t ws_size,
                              hipStream_t stream) {
  const float*  x     = (const float*)d_in[0];
  const int8_t* w     = (const int8_t*)d_in[1];
  const float*  scale = (const float*)d_in[2];
  const float*  bias  = (const float*)d_in[3];
  const float*  act   = (const float*)d_in[4];
  float* out = (float*)d_out;

  dim3 grid(Ndim / BN, Mdim / BM);
  const size_t qbytes = (size_t)Mdim * Kdim;

  if (ws_size >= qbytes) {
    const int nblk = (Mdim * Kdim / 4) / 256;
    quantize_kernel<<<nblk, 256, 0, stream>>>(x, (uint32_t*)d_ws, act);
    qgemm_wmma_async<<<grid, 256, 0, stream>>>(
        (const int8_t*)d_ws, w, scale, bias, act, out);
  } else {
    qgemm_wmma_fused<<<grid, 256, 0, stream>>>(x, w, scale, bias, act, out);
  }
}
